// LatticeBERT_46488726012522
// MI455X (gfx1250) — matrix operand order repaired and checked by reading, not compiled, not run on hardware
//
#include <hip/hip_runtime.h>
#include <hip/hip_bf16.h>

#define BATCH 8
#define SEQ   1024
#define HID   1024
#define NH    16
#define HD    64
#define MROWS (BATCH*SEQ)   // 8192

#define USE_TDM 1

#if defined(__HIP_DEVICE_COMPILE__) && defined(__has_builtin)
#  if __has_builtin(__builtin_amdgcn_cvt_pk_bf16_f32)
#    define HAVE_CVT_PK_BF16 1
#  endif
#endif

typedef __attribute__((ext_vector_type(16))) __bf16 v16bf;
typedef __attribute__((ext_vector_type(8)))  float  v8f;
typedef unsigned int u32x4 __attribute__((ext_vector_type(4)));
typedef int          i32x8 __attribute__((ext_vector_type(8)));
typedef int          i32x4 __attribute__((ext_vector_type(4)));

union Frag16 {
    v16bf v;
    unsigned short u[16];
    uint4 q[2];
};

__device__ __forceinline__ unsigned pack2bf(float lo, float hi) {
#ifdef HAVE_CVT_PK_BF16
    auto t = __builtin_amdgcn_cvt_pk_bf16_f32(lo, hi);   // v_cvt_pk_bf16_f32
    unsigned r;
    __builtin_memcpy(&r, &t, 4);
    return r;
#else
    unsigned x = __float_as_uint(lo);
    x += 0x7FFFu + ((x >> 16) & 1u);
    unsigned y = __float_as_uint(hi);
    y += 0x7FFFu + ((y >> 16) & 1u);
    return (x >> 16) | (y & 0xFFFF0000u);
#endif
}
__device__ __forceinline__ unsigned short f2bf(float f) {
#ifdef HAVE_CVT_PK_BF16
    return (unsigned short)(pack2bf(f, 0.0f) & 0xFFFFu);
#else
    unsigned int x = __float_as_uint(f);
    x += 0x7FFFu + ((x >> 16) & 1u);      // round-to-nearest-even
    return (unsigned short)(x >> 16);
#endif
}

__device__ __forceinline__ v8f wmma_bf16(const Frag16& a, const Frag16& b, v8f c) {
    return __builtin_amdgcn_wmma_f32_16x16x32_bf16(false, a.v, false, b.v,
                                                   (short)0, c, false, false);
}

#if USE_TDM
// 2-D TDM tile load: bf16 elements, LDS rows padded by 8 bf16 (4 dwords) every
// 64 bf16 (32 dwords). D# layout per CDNA5 ISA.
// This toolchain exposes the 6-arg builtin: (g0, g1, g2, g3, g4, cpol).
__device__ __forceinline__ void tdm_load_tile_2d(unsigned ldsByteOff,
                                                 const unsigned short* gptr,
                                                 unsigned tensor_d0, unsigned tensor_d1,
                                                 unsigned stride0,
                                                 unsigned tile_d0, unsigned tile_d1) {
    unsigned long long ga = (unsigned long long)(uintptr_t)gptr;
    u32x4 g0;
    g0[0] = 1u;                                           // count=1, user D#
    g0[1] = ldsByteOff;                                   // lds_addr
    g0[2] = (unsigned)(ga & 0xFFFFFFFFu);                 // global_addr[31:0]
    g0[3] = (unsigned)((ga >> 32) & 0x1FFFFFFu) | (2u << 30);  // addr[56:32] | type=2
    i32x8 g1;
    g1[0] = (1 << 16)       // data_size = 2 bytes
          | (1 << 20)       // pad_enable
          | (4 << 22)       // pad_interval: 32 dwords (64 bf16)
          | (3 << 25);      // pad_amount:   4 dwords (8 bf16)
    g1[1] = (int)(tensor_d0 << 16);                             // tensor_dim0[15:0]
    g1[2] = (int)((tensor_d0 >> 16) | (tensor_d1 << 16));       // d0 hi | d1 lo
    g1[3] = (int)((tensor_d1 >> 16) | (tile_d0 << 16));         // d1 hi | tile_dim0
    g1[4] = (int)tile_d1;                                       // tile_dim1 (tile_dim2=0)
    g1[5] = (int)stride0;                                       // tensor_dim0_stride lo
    g1[6] = 0;
    g1[7] = 0;
    i32x4 z4;
    z4[0] = 0; z4[1] = 0; z4[2] = 0; z4[3] = 0;
    i32x8 z8;
    z8[0] = 0; z8[1] = 0; z8[2] = 0; z8[3] = 0;
    z8[4] = 0; z8[5] = 0; z8[6] = 0; z8[7] = 0;
    __builtin_amdgcn_tensor_load_to_lds(g0, g1, z4, z4, z8, 0);
}
#endif

// ---------------------------------------------------------------- utilities
__global__ void zero_f32_kernel(float* __restrict__ p, size_t n) {
    size_t i = (size_t)blockIdx.x * blockDim.x + threadIdx.x;
    size_t stride = (size_t)gridDim.x * blockDim.x;
    for (; i < n; i += stride) p[i] = 0.0f;
}

__global__ void scatter_bias_kernel(const int* __restrict__ eb,
                                    const int* __restrict__ ech,
                                    const int* __restrict__ ew,
                                    float* __restrict__ bias, int n) {
    int i = blockIdx.x * blockDim.x + threadIdx.x;
    if (i < n) {
        size_t idx = (((size_t)eb[i] * SEQ) + ech[i]) * SEQ + ew[i];
        atomicAdd(&bias[idx], 1.5f);
    }
}

// ---------------------------------------------------------------- QKV GEMM
// C[M=8192, N=1024] = X @ W + b. Q/K stored bf16 head-major [b][h][s][d];
// V stored TRANSPOSED [b][h][d][s] so attention V fragments read contiguously.
// blockIdx.z selects Q(0)/K(1)/V(2); Q is pre-scaled by 1/sqrt(hd)=0.125.
#define GBM 64
#define GBN 128
#define GBK 32
#define APAD 8

__global__ __launch_bounds__(256)
void qkv_gemm_kernel(const float* __restrict__ X,
                     const float* __restrict__ Wq, const float* __restrict__ bq,
                     const float* __restrict__ Wk, const float* __restrict__ bk,
                     const float* __restrict__ Wv, const float* __restrict__ bv,
                     unsigned short* __restrict__ Qo,
                     unsigned short* __restrict__ Ko,
                     unsigned short* __restrict__ Vo) {
    const int which = blockIdx.z;
    const float* W    = (which == 0) ? Wq : (which == 1) ? Wk : Wv;
    const float* bias = (which == 0) ? bq : (which == 1) ? bk : bv;
    unsigned short* Out = (which == 0) ? Qo : (which == 1) ? Ko : Vo;
    const float scale = (which == 0) ? 0.125f : 1.0f;

    __shared__ __align__(16) unsigned short As[GBM][GBK + APAD];   // [m][k]
    __shared__ __align__(16) unsigned short Bs[GBN][GBK + APAD];   // [n][k] (transposed)

    const int tid  = threadIdx.x;
    const int m0   = blockIdx.y * GBM;
    const int n0   = blockIdx.x * GBN;
    const int wave = tid >> 5;
    const int lane = tid & 31;
    const int half = lane >> 4;
    const int lr   = lane & 15;
    const int wrow = (wave >> 2) * 32;  // 0 / 32
    const int wcol = (wave & 3) * 32;   // 0..96

    v8f acc[2][2];
    #pragma unroll
    for (int ms = 0; ms < 2; ++ms)
        #pragma unroll
        for (int ns = 0; ns < 2; ++ns)
            #pragma unroll
            for (int r = 0; r < 8; ++r) acc[ms][ns][r] = 0.0f;

    for (int k0 = 0; k0 < HID; k0 += GBK) {
        __syncthreads();
        // stage A tile 64x32: f32 -> packed bf16, b64 LDS stores
        for (int i = tid; i < GBM * GBK / 4; i += 256) {
            int r = i >> 3, c = (i & 7) * 4;
            float4 f = *(const float4*)(X + (size_t)(m0 + r) * HID + k0 + c);
            uint2 p;
            p.x = pack2bf(f.x, f.y);
            p.y = pack2bf(f.z, f.w);
            *(uint2*)&As[r][c] = p;
        }
        // stage B tile 32x128 transposed -> Bs[n][k], packed b32 LDS stores
        for (int i = tid; i < (GBK / 2) * (GBN / 4); i += 256) {
            int kp = (i >> 5) * 2, c = (i & 31) * 4;
            float4 f0 = *(const float4*)(W + (size_t)(k0 + kp)     * HID + n0 + c);
            float4 f1 = *(const float4*)(W + (size_t)(k0 + kp + 1) * HID + n0 + c);
            *(unsigned*)&Bs[c + 0][kp] = pack2bf(f0.x, f1.x);
            *(unsigned*)&Bs[c + 1][kp] = pack2bf(f0.y, f1.y);
            *(unsigned*)&Bs[c + 2][kp] = pack2bf(f0.z, f1.z);
            *(unsigned*)&Bs[c + 3][kp] = pack2bf(f0.w, f1.w);
        }
        if (k0 + GBK < HID) {
            __builtin_prefetch(X + (size_t)(m0 + (tid >> 2)) * HID + k0 + GBK, 0, 3);
            __builtin_prefetch(W + (size_t)(k0 + GBK + (tid >> 3)) * HID + n0, 0, 3);
        }
        __syncthreads();

        Frag16 a[2], b[2];
        #pragma unroll
        for (int ms = 0; ms < 2; ++ms) {
            int row = wrow + ms * 16 + lr;
            a[ms].q[0] = *(const uint4*)&As[row][half * 8];
            a[ms].q[1] = *(const uint4*)&As[row][16 + half * 8];
        }
        #pragma unroll
        for (int ns = 0; ns < 2; ++ns) {
            int col = wcol + ns * 16 + lr;
            b[ns].q[0] = *(const uint4*)&Bs[col][half * 8];
            b[ns].q[1] = *(const uint4*)&Bs[col][16 + half * 8];
        }
        #pragma unroll
        for (int ms = 0; ms < 2; ++ms)
            #pragma unroll
            for (int ns = 0; ns < 2; ++ns)
                acc[ms][ns] = wmma_bf16(a[ms], b[ns], acc[ms][ns]);
    }

    #pragma unroll
    for (int ms = 0; ms < 2; ++ms)
        #pragma unroll
        for (int ns = 0; ns < 2; ++ns) {
            int coln = n0 + wcol + ns * 16 + lr;
            float bv_ = bias[coln];
            int h = coln >> 6, d = coln & 63;
            #pragma unroll
            for (int r = 0; r < 8; ++r) {
                int grow = m0 + wrow + ms * 16 + half * 8 + r;
                int bb = grow >> 10, s = grow & 1023;
                float val = (acc[ms][ns][r] + bv_) * scale;
                size_t idx = (which == 2)
                    ? ((((size_t)bb * NH + h) * HD + d) * SEQ) + s   // V: [b][h][d][s]
                    : ((((size_t)bb * NH + h) * SEQ + s) * HD) + d;  // Q,K: [b][h][s][d]
                Out[idx] = f2bf(val);
            }
        }
}

// ---------------------------------------------------------------- attention
// Flash-attention. Block = (b, h, 64-row q-tile); 4 waves, each owns 16 rows.
// K/V tiles staged to LDS via the Tensor Data Mover (TDM), rows padded +8 bf16.
__global__ __launch_bounds__(128)
void attn_kernel(const unsigned short* __restrict__ Qb,
                 const unsigned short* __restrict__ Kb,
                 const unsigned short* __restrict__ Vb,   // [b][h][d][s]
                 const float* __restrict__ bias,
                 const int* __restrict__ mask,
                 unsigned short* __restrict__ Ctx) {
    const int qt   = blockIdx.x;   // 0..15
    const int h    = blockIdx.y;   // 0..15
    const int b    = blockIdx.z;   // 0..7
    const int tid  = threadIdx.x;
    const int wave = tid >> 5;
    const int lane = tid & 31;
    const int half = lane >> 4;
    const int lr   = lane & 15;

    __shared__ __align__(16) unsigned short Ks[64][HD + 8];    // [kv][d]
    __shared__ __align__(16) unsigned short VsT[HD][64 + 8];   // [d][kv]
    __shared__ __align__(16) unsigned short Ps[4][16][64 + 8]; // per-wave P

    const size_t headBase = (((size_t)b * NH + h) * SEQ) * HD;  // == (b*NH+h)*HD*SEQ
    const int q0 = qt * 64 + wave * 16;

    // Q fragments (A operand), loaded once: 16x64 bf16 = 2 k-steps, b128 loads
    Frag16 aQ[2];
    {
        const unsigned short* qp = Qb + headBase + (size_t)(q0 + lr) * HD;
        #pragma unroll
        for (int ks = 0; ks < 2; ++ks) {
            aQ[ks].q[0] = *(const uint4*)(qp + ks * 32 + half * 8);
            aQ[ks].q[1] = *(const uint4*)(qp + ks * 32 + 16 + half * 8);
        }
    }

    float mrow[8], lrow[8];
    v8f accO[4];
    #pragma unroll
    for (int r = 0; r < 8; ++r) { mrow[r] = -1e30f; lrow[r] = 0.0f; }
    #pragma unroll
    for (int n = 0; n < 4; ++n)
        #pragma unroll
        for (int r = 0; r < 8; ++r) accO[n][r] = 0.0f;

    for (int kv0 = 0; kv0 < SEQ; kv0 += 64) {
        __syncthreads();   // previous tile fully consumed
#if USE_TDM
        if (wave == 0) {   // TDM ignores EXEC; issue from one wave only
            // K tile: tensor [SEQ][HD], tile 64 rows x 64 cols -> Ks[kv][d]
            tdm_load_tile_2d((unsigned)(uintptr_t)&Ks[0][0],
                             Kb + headBase + (size_t)kv0 * HD,
                             HD, SEQ, HD, HD, 64);
            // V tile: tensor [HD][SEQ], tile 64 rows x 64 cols -> VsT[d][kv]
            tdm_load_tile_2d((unsigned)(uintptr_t)&VsT[0][0],
                             Vb + headBase + kv0,
                             SEQ, HD, SEQ, 64, HD);
            __builtin_amdgcn_s_wait_tensorcnt(0);
        }
#else
        for (int i = tid; i < 512; i += 128) {
            int r = i >> 3, c = (i & 7) * 8;
            *(uint4*)&Ks[r][c]  = *(const uint4*)(Kb + headBase + (size_t)(kv0 + r) * HD + c);
            *(uint4*)&VsT[r][c] = *(const uint4*)(Vb + headBase + (size_t)r * SEQ + kv0 + c);
        }
#endif
        __syncthreads();

        // scores: 4 tiles of 16 kv columns
        v8f sc[4];
        #pragma unroll
        for (int t = 0; t < 4; ++t) {
            v8f s;
            #pragma unroll
            for (int r = 0; r < 8; ++r) s[r] = 0.0f;
            #pragma unroll
            for (int ks = 0; ks < 2; ++ks) {
                Frag16 bK;   // B = K^T : element (d, n) = K[n][d], contiguous in d
                int kvcol = t * 16 + lr;
                bK.q[0] = *(const uint4*)&Ks[kvcol][ks * 32 + half * 8];
                bK.q[1] = *(const uint4*)&Ks[kvcol][ks * 32 + 16 + half * 8];
                s = wmma_bf16(aQ[ks], bK, s);
            }
            // bias + attention mask in C layout
            int gcol = kv0 + t * 16 + lr;
            float madd = (mask[b * SEQ + gcol] == 0) ? -1e9f : 0.0f;
            const float* bp = bias + ((size_t)b * SEQ) * SEQ + gcol;
            #pragma unroll
            for (int r = 0; r < 8; ++r) {
                int grow = qt * 64 + wave * 16 + half * 8 + r;
                s[r] += madd + bp[(size_t)grow * SEQ];
            }
            sc[t] = s;
        }

        // online softmax (row reductions across the 16-lane half)
        float alpha[8];
        #pragma unroll
        for (int r = 0; r < 8; ++r) {
            float tmax = fmaxf(fmaxf(sc[0][r], sc[1][r]), fmaxf(sc[2][r], sc[3][r]));
            #pragma unroll
            for (int off = 1; off < 16; off <<= 1)
                tmax = fmaxf(tmax, __shfl_xor(tmax, off, 32));
            float mnew = fmaxf(mrow[r], tmax);
            alpha[r] = __expf(mrow[r] - mnew);
            mrow[r] = mnew;
            float ls = 0.0f;
            #pragma unroll
            for (int t = 0; t < 4; ++t) {
                float p = __expf(sc[t][r] - mnew);
                sc[t][r] = p;
                ls += p;
            }
            #pragma unroll
            for (int off = 1; off < 16; off <<= 1)
                ls += __shfl_xor(ls, off, 32);
            lrow[r] = lrow[r] * alpha[r] + ls;
        }
        #pragma unroll
        for (int n = 0; n < 4; ++n)
            #pragma unroll
            for (int r = 0; r < 8; ++r) accO[n][r] *= alpha[r];

        // C-layout -> A-layout via per-wave LDS round trip
        #pragma unroll
        for (int t = 0; t < 4; ++t)
            #pragma unroll
            for (int r = 0; r < 8; ++r)
                Ps[wave][half * 8 + r][t * 16 + lr] = f2bf(sc[t][r]);

        // O += P @ V  (k-dim = 64 kv, 2 k-steps; 4 d-tiles), all b128 operand loads
        #pragma unroll
        for (int ks = 0; ks < 2; ++ks) {
            Frag16 aP;
            aP.q[0] = *(const uint4*)&Ps[wave][lr][ks * 32 + half * 8];
            aP.q[1] = *(const uint4*)&Ps[wave][lr][ks * 32 + 16 + half * 8];
            #pragma unroll
            for (int n = 0; n < 4; ++n) {
                Frag16 bV;   // B = V : element (kv, d) = VsT[d][kv], contiguous in kv
                bV.q[0] = *(const uint4*)&VsT[n * 16 + lr][ks * 32 + half * 8];
                bV.q[1] = *(const uint4*)&VsT[n * 16 + lr][ks * 32 + 16 + half * 8];
                accO[n] = wmma_bf16(aP, bV, accO[n]);
            }
        }
    }

    // normalize and store context bf16  [b][s][h*64+d]
    #pragma unroll
    for (int n = 0; n < 4; ++n) {
        int d = n * 16 + lr;
        #pragma unroll
        for (int r = 0; r < 8; ++r) {
            int grow = qt * 64 + wave * 16 + half * 8 + r;
            float val = accO[n][r] / lrow[r];
            Ctx[((size_t)b * SEQ + grow) * HID + h * HD + d] = f2bf(val);
        }
    }
}

// ---------------------------------------------------------------- output GEMM
__global__ __launch_bounds__(256)
void out_gemm_kernel(const unsigned short* __restrict__ Ctx,
                     const float* __restrict__ Wo,
                     const float* __restrict__ bo,
                     float* __restrict__ Y) {
    __shared__ __align__(16) unsigned short As[GBM][GBK + APAD];   // [m][k]
    __shared__ __align__(16) unsigned short Bs[GBN][GBK + APAD];   // [n][k]

    const int tid  = threadIdx.x;
    const int m0   = blockIdx.y * GBM;
    const int n0   = blockIdx.x * GBN;
    const int wave = tid >> 5;
    const int lane = tid & 31;
    const int half = lane >> 4;
    const int lr   = lane & 15;
    const int wrow = (wave >> 2) * 32;
    const int wcol = (wave & 3) * 32;

    v8f acc[2][2];
    #pragma unroll
    for (int ms = 0; ms < 2; ++ms)
        #pragma unroll
        for (int ns = 0; ns < 2; ++ns)
            #pragma unroll
            for (int r = 0; r < 8; ++r) acc[ms][ns][r] = 0.0f;

    for (int k0 = 0; k0 < HID; k0 += GBK) {
        __syncthreads();
        // stage A tile (already bf16): one b128 per thread
        {
            int r = tid >> 2, c = (tid & 3) * 8;
            *(uint4*)&As[r][c] = *(const uint4*)(Ctx + (size_t)(m0 + r) * HID + k0 + c);
        }
        // stage B tile transposed, packed b32 stores
        for (int i = tid; i < (GBK / 2) * (GBN / 4); i += 256) {
            int kp = (i >> 5) * 2, c = (i & 31) * 4;
            float4 f0 = *(const float4*)(Wo + (size_t)(k0 + kp)     * HID + n0 + c);
            float4 f1 = *(const float4*)(Wo + (size_t)(k0 + kp + 1) * HID + n0 + c);
            *(unsigned*)&Bs[c + 0][kp] = pack2bf(f0.x, f1.x);
            *(unsigned*)&Bs[c + 1][kp] = pack2bf(f0.y, f1.y);
            *(unsigned*)&Bs[c + 2][kp] = pack2bf(f0.z, f1.z);
            *(unsigned*)&Bs[c + 3][kp] = pack2bf(f0.w, f1.w);
        }
        if (k0 + GBK < HID) {
            __builtin_prefetch(Ctx + (size_t)(m0 + (tid >> 2)) * HID + k0 + GBK, 0, 3);
            __builtin_prefetch(Wo + (size_t)(k0 + GBK + (tid >> 3)) * HID + n0, 0, 3);
        }
        __syncthreads();

        Frag16 a[2], b[2];
        #pragma unroll
        for (int ms = 0; ms < 2; ++ms) {
            int row = wrow + ms * 16 + lr;
            a[ms].q[0] = *(const uint4*)&As[row][half * 8];
            a[ms].q[1] = *(const uint4*)&As[row][16 + half * 8];
        }
        #pragma unroll
        for (int ns = 0; ns < 2; ++ns) {
            int col = wcol + ns * 16 + lr;
            b[ns].q[0] = *(const uint4*)&Bs[col][half * 8];
            b[ns].q[1] = *(const uint4*)&Bs[col][16 + half * 8];
        }
        #pragma unroll
        for (int ms = 0; ms < 2; ++ms)
            #pragma unroll
            for (int ns = 0; ns < 2; ++ns)
                acc[ms][ns] = wmma_bf16(a[ms], b[ns], acc[ms][ns]);
    }

    #pragma unroll
    for (int ms = 0; ms < 2; ++ms)
        #pragma unroll
        for (int ns = 0; ns < 2; ++ns) {
            int coln = n0 + wcol + ns * 16 + lr;
            float bv_ = bo[coln];
            #pragma unroll
            for (int r = 0; r < 8; ++r) {
                int grow = m0 + wrow + ms * 16 + half * 8 + r;
                Y[(size_t)grow * HID + coln] = acc[ms][ns][r] + bv_;
            }
        }
}

// ---------------------------------------------------------------- launch
extern "C" void kernel_launch(void* const* d_in, const int* in_sizes, int n_in,
                              void* d_out, int out_size, void* d_ws, size_t ws_size,
                              hipStream_t stream) {
    const float* X    = (const float*)d_in[0];
    const float* Wq   = (const float*)d_in[1];
    const float* bq   = (const float*)d_in[2];
    const float* Wk   = (const float*)d_in[3];
    const float* bk   = (const float*)d_in[4];
    const float* Wv   = (const float*)d_in[5];
    const float* bv   = (const float*)d_in[6];
    const float* Wo   = (const float*)d_in[7];
    const float* bo   = (const float*)d_in[8];
    const int*   mask = (const int*)d_in[9];
    const int*   eb   = (const int*)d_in[10];
    const int*   ech  = (const int*)d_in[11];
    const int*   ew   = (const int*)d_in[12];
    float* Y = (float*)d_out;

    char* ws = (char*)d_ws;
    unsigned short* Qb  = (unsigned short*)(ws);
    unsigned short* Kb  = (unsigned short*)(ws + (size_t)16 * 1024 * 1024);
    unsigned short* Vb  = (unsigned short*)(ws + (size_t)32 * 1024 * 1024);
    float*          Bia = (float*)(ws + (size_t)48 * 1024 * 1024);
    unsigned short* Ctx = (unsigned short*)(ws + (size_t)80 * 1024 * 1024);

    const size_t biasN = (size_t)BATCH * SEQ * SEQ;
    zero_f32_kernel<<<8192, 256, 0, stream>>>(Bia, biasN);
    scatter_bias_kernel<<<(16384 + 255) / 256, 256, 0, stream>>>(eb, ech, ew, Bia, 16384);

    dim3 g1(HID / GBN, MROWS / GBM, 3);
    qkv_gemm_kernel<<<g1, 256, 0, stream>>>(X, Wq, bq, Wk, bk, Wv, bv, Qb, Kb, Vb);

    dim3 g2(SEQ / 64, NH, BATCH);
    attn_kernel<<<g2, 128, 0, stream>>>(Qb, Kb, Vb, Bia, mask, Ctx);

    dim3 g3(HID / GBN, MROWS / GBM);
    out_gemm_kernel<<<g3, 256, 0, stream>>>(Ctx, Wo, bo, Y);
}
